// RelationalReasoningModuleNoTranspose_26388279067058
// MI455X (gfx1250) — matrix operand order repaired and checked by reading, not compiled, not run on hardware
//
#include <hip/hip_runtime.h>
#include <hip/hip_bf16.h>

// ---------------------------------------------------------------------------
// RelationalReasoningModule, restructured:
//  k1: per batch b:  A = W1[:,0:64]@X, Bm = W1[:,64:128]@X  (WMMA f16->f32,
//      f32 results kept), c = W1[:,128:192]@emb + b1
//  k2: hsum_partial = sum_{i<j} relu(A[:,i]+Bm[:,j]+c)  (VALU, B in LDS via
//      global_load_async_to_lds_b128 / ASYNCcnt)
//  k3: reduce partials; out = W2 @ hsum + R*b2          (WMMA f32 16x16x4)
// ---------------------------------------------------------------------------

typedef __attribute__((ext_vector_type(16))) _Float16 v16h;
typedef __attribute__((ext_vector_type(8)))  _Float16 v8h;
typedef __attribute__((ext_vector_type(8)))  float    v8f;
typedef __attribute__((ext_vector_type(4)))  float    v4f;
typedef __attribute__((ext_vector_type(2)))  float    v2f;

#define BATCH 16
#define DD    64     // channels per node
#define NN    256    // nodes (16x16)
#define HID   128
#define INW   192    // W1 row length
#define OUTC  64
#define SLICES 16

// Build a v16h A/B fragment from two 16-byte LDS chunks (ds_load_b128 x2)
__device__ inline v16h ld_frag2(const _Float16* p0, const _Float16* p1) {
  v8h a = *(const v8h*)p0;
  v8h b = *(const v8h*)p1;
  v16h r;
#pragma unroll
  for (int i = 0; i < 8; ++i) { r[i] = a[i]; r[i + 8] = b[i]; }
  return r;
}

// ---------------------------------------------------------------------------
// Kernel 1: per-batch GEMMs A=W1a@X, B=W1b@X via v_wmma_f32_16x16x32_f16,
//           results kept in f32; plus c = W1e@emb + b1.
//           grid=16 (batch), block=256 (8 waves).
// ---------------------------------------------------------------------------
__global__ __launch_bounds__(256) void k1_prep(
    const float* __restrict__ x, const float* __restrict__ emb,
    const float* __restrict__ W1, const float* __restrict__ b1,
    float* __restrict__ Aws, float* __restrict__ Bws,
    float* __restrict__ cws) {
  __shared__ __align__(16) _Float16 xT[NN * DD];     // X transposed [n][d], 32KB
  __shared__ __align__(16) _Float16 w1l[HID * 128];  // W1[:,0:128] f16,   32KB

  const int b   = blockIdx.x;
  const int tid = threadIdx.x;

  // Load x[b] (64x256 f32, row=d) transposed into LDS as f16 [n][d]
  const float* xb = x + (size_t)b * DD * NN;
  for (int d = 0; d < DD; ++d) {
    float v = xb[d * NN + tid];            // coalesced across threads
    xT[tid * DD + d] = (_Float16)v;
  }
  // Load W1 columns 0..127 as f16
  for (int idx = tid; idx < HID * 128; idx += 256) {
    int h = idx >> 7, k = idx & 127;
    w1l[idx] = (_Float16)W1[h * INW + k];
  }
  // c[h] = b1[h] + sum_e W1[h][128+e]*emb[b][e]
  if (tid < HID) {
    float acc = b1[tid];
    const float* wr = W1 + tid * INW + 128;
    const float* e  = emb + b * 64;
#pragma unroll 8
    for (int q = 0; q < 64; ++q) acc += wr[q] * e[q];
    cws[b * HID + tid] = acc;
  }
  __syncthreads();

  const int w = tid >> 5, l = tid & 31;
  const int m0 = w * 16;           // each wave owns one 16-row tile of HID
  const int lr = l & 15, lh = l >> 4;

  // A-matrix fragments (W1 rows), layout per ISA 16-bit A 16x32:
  // lanes 0-15: halves0-7 = K0..7, halves8-15 = K16..23 (lanes16-31 +8)
  v16h aA[2], aB[2];
  {
    const _Float16* base = &w1l[(m0 + lr) * 128];
#pragma unroll
    for (int kk = 0; kk < 2; ++kk) {
      aA[kk] = ld_frag2(base + kk * 32 + lh * 8,
                        base + kk * 32 + lh * 8 + 16);
      aB[kk] = ld_frag2(base + 64 + kk * 32 + lh * 8,
                        base + 64 + kk * 32 + lh * 8 + 16);
    }
  }

  for (int n0t = 0; n0t < 16; ++n0t) {
    const int n = n0t * 16 + lr;
    // B-matrix fragment (X cols), 16-bit B 32x16: halves0-7 = K(l/16)*8+0..7, halves8-15 = K16+(l/16)*8+0..7
    v16h bf[2];
#pragma unroll
    for (int kk = 0; kk < 2; ++kk) {
      const _Float16* bb = &xT[n * DD + kk * 32 + lh * 8];
      bf[kk] = ld_frag2(bb, bb + 16);
    }
    v8f cA = {}, cB = {};
    cA = __builtin_amdgcn_wmma_f32_16x16x32_f16(false, aA[0], false, bf[0],
                                                (short)0, cA, false, false);
    cA = __builtin_amdgcn_wmma_f32_16x16x32_f16(false, aA[1], false, bf[1],
                                                (short)0, cA, false, false);
    cB = __builtin_amdgcn_wmma_f32_16x16x32_f16(false, aB[0], false, bf[0],
                                                (short)0, cB, false, false);
    cB = __builtin_amdgcn_wmma_f32_16x16x32_f16(false, aB[1], false, bf[1],
                                                (short)0, cB, false, false);
    // D layout: vgpr g, lanes0-15 -> M=g, lanes16-31 -> M=8+g.
    // Store f32 [b][n][m]: 8 consecutive m -> one 32B store (2x b128).
    size_t off = (size_t)b * NN * HID + (size_t)n * HID + m0 + lh * 8;
    *(v8f*)(Aws + off) = cA;
    *(v8f*)(Bws + off) = cB;
  }
}

// ---------------------------------------------------------------------------
// Kernel 2: pairwise relu-sum. grid=(SLICES, BATCH), block=256 (8 waves).
// Wave handles i with i%SLICES==s, (i/SLICES)%8==w; lane owns 4 channels.
// Bm[b] (128KB f32) is pulled into LDS with GLOBAL_LOAD_ASYNC_TO_LDS_B128
// (ASYNCcnt path, no VGPR staging); deterministic per-wave partials out.
// ---------------------------------------------------------------------------
__global__ __launch_bounds__(256) void k2_pair(
    const float* __restrict__ Aws, const float* __restrict__ Bws,
    const float* __restrict__ cws, float* __restrict__ Pws) {
  __shared__ __align__(16) float Bl[NN * HID];  // 128KB (of 320KB/WGP)

  const int s = blockIdx.x, b = blockIdx.y;
  const int tid = threadIdx.x;

  // Async copy Bws[b] -> LDS: 8192 x 16B, 256 lanes x 32 iterations.
  // GVS addressing: global = SADDR + VADDR; LDS dest = VGPR (wave-relative,
  // low 32 bits of the generic pointer per flat->LDS truncation).
  {
    const float* src = Bws + (size_t)b * NN * HID;
    unsigned ldsbase = (unsigned)(unsigned long long)(const void*)&Bl[0];
    unsigned byteoff = (unsigned)tid * 16u;
#pragma unroll 4
    for (int it = 0; it < 32; ++it) {
      asm volatile("global_load_async_to_lds_b128 %0, %1, %2"
                   :
                   : "v"(ldsbase + byteoff), "v"(byteoff), "s"(src)
                   : "memory");
      byteoff += 256u * 16u;
    }
    asm volatile("s_wait_asynccnt 0x0" ::: "memory");
  }
  __syncthreads();

  const int w = tid >> 5, l = tid & 31;
  const int ch = l * 4;
  const v4f c4 = *(const v4f*)(cws + b * HID + ch);
  v4f acc = {0.f, 0.f, 0.f, 0.f};

  for (int i = s + SLICES * w; i < NN - 1; i += SLICES * 8) {
    v4f a4 = *(const v4f*)(Aws + (size_t)b * NN * HID + (size_t)i * HID + ch);
    const float a0 = a4[0] + c4[0];
    const float a1 = a4[1] + c4[1];
    const float a2 = a4[2] + c4[2];
    const float a3 = a4[3] + c4[3];
    for (int j = i + 1; j < NN; ++j) {
      v4f b4 = *(const v4f*)&Bl[j * HID + ch];     // ds_load_b128
      acc[0] += fmaxf(a0 + b4[0], 0.f);
      acc[1] += fmaxf(a1 + b4[1], 0.f);
      acc[2] += fmaxf(a2 + b4[2], 0.f);
      acc[3] += fmaxf(a3 + b4[3], 0.f);
    }
  }
  // Per-(slice,wave) partial: unique address, no atomics -> deterministic.
  float* p = Pws + ((size_t)b * 128 + (size_t)(s * 8 + w)) * HID + ch;
  *(v4f*)p = acc;
}

// ---------------------------------------------------------------------------
// Kernel 3: reduce partials to hsum[b][h], then out = W2@hsum + R*b2 using
// exact-f32 v_wmma_f32_16x16x4_f32 (M=64 o, N=16 b, K=128). grid=1, block=128.
// ---------------------------------------------------------------------------
__global__ __launch_bounds__(128) void k3_out(
    const float* __restrict__ Pws, const float* __restrict__ W2,
    const float* __restrict__ b2, float* __restrict__ out) {
  __shared__ float hs[BATCH * HID];  // 8KB

  const int tid = threadIdx.x;
  for (int idx = tid; idx < BATCH * HID; idx += 128) {
    const int b = idx >> 7, h = idx & 127;
    const float* p = Pws + (size_t)b * 128 * HID + h;
    float sum = 0.f;
#pragma unroll 4
    for (int k = 0; k < 128; ++k) sum += p[k * HID];
    hs[idx] = sum;
  }
  __syncthreads();

  const int w = tid >> 5, l = tid & 31;
  const int m0 = w * 16, lr = l & 15, lh = l >> 4;
  v8f d = {};
  for (int k0 = 0; k0 < HID; k0 += 4) {
    // f32 A 16x4: v0 -> K=k0+(l/16)*2, v1 -> K+1  (consecutive pair)
    v2f a = *(const v2f*)(W2 + (size_t)(m0 + lr) * HID + k0 + lh * 2);
    // f32 B 4x16: v0 -> K=k0+(l/16), v1 -> K=k0+2+(l/16); N = lane%16 = batch
    v2f bb;
    bb[0] = hs[lr * HID + k0 + lh];
    bb[1] = hs[lr * HID + k0 + 2 + lh];
    d = __builtin_amdgcn_wmma_f32_16x16x4_f32(false, a, false, bb,
                                              (short)0, d, false, false);
  }
#pragma unroll
  for (int g = 0; g < 8; ++g) {
    const int o = m0 + lh * 8 + g;
    out[lr * OUTC + o] = d[g] + 32640.0f * b2[o];  // R = 256*255/2
  }
}

// ---------------------------------------------------------------------------
extern "C" void kernel_launch(void* const* d_in, const int* in_sizes, int n_in,
                              void* d_out, int out_size, void* d_ws,
                              size_t ws_size, hipStream_t stream) {
  const float* x   = (const float*)d_in[0];
  const float* emb = (const float*)d_in[1];
  const float* W1  = (const float*)d_in[2];
  const float* b1  = (const float*)d_in[3];
  const float* W2  = (const float*)d_in[4];
  const float* b2  = (const float*)d_in[5];
  float* out = (float*)d_out;

  // Workspace layout (bytes):
  //  [0, 2M)        Aws  f32 [16][256][128]
  //  [2M, 4M)       Bws  f32 [16][256][128]
  //  [4M, 4M+8K)    cws  f32 [16][128]
  //  [4M+8K, +1M)   Pws  f32 [16][128 slices*waves][128]
  char* ws = (char*)d_ws;
  float* Aws = (float*)(ws);
  float* Bws = (float*)(ws + (2u << 20));
  float* cws = (float*)(ws + (4u << 20));
  float* Pws = (float*)(ws + (4u << 20) + 8192);

  k1_prep<<<BATCH, 256, 0, stream>>>(x, emb, W1, b1, Aws, Bws, cws);
  k2_pair<<<dim3(SLICES, BATCH), 256, 0, stream>>>(Aws, Bws, cws, Pws);
  k3_out<<<1, 128, 0, stream>>>(Pws, W2, b2, out);
}
